// VQEmbedding_1065151889564
// MI455X (gfx1250) — matrix operand order repaired and checked by reading, not compiled, not run on hardware
//
#include <hip/hip_runtime.h>
#include <stdint.h>

typedef __attribute__((ext_vector_type(16))) _Float16 v16h;
typedef __attribute__((ext_vector_type(8)))  _Float16 v8h;
typedef __attribute__((ext_vector_type(8)))  float    v8f;
typedef __attribute__((ext_vector_type(4)))  float    v4f;
typedef __attribute__((ext_vector_type(2)))  float    v2f;

#define K_CODES 1024
#define D_DIM   256
#define N_PIX   65536                  // B*H*W
#define NT      (K_CODES / 16)         // 64 code tiles
#define LDS_STRIDE 264                 // halves per staged code row (16B-aligned, bank-spread)

// =======================================================================
// prep: embedding f32 -> f16 (packed [K][256]) + per-code ||e||^2
// one wave per code row
// =======================================================================
__global__ __launch_bounds__(32)
void vq_prep_kernel(const float* __restrict__ emb,
                    _Float16* __restrict__ eh, float* __restrict__ en) {
  const int k = blockIdx.x;
  const int lane = threadIdx.x;
  const float* src = emb + k * D_DIM;
  _Float16*    dst = eh  + k * D_DIM;
  float s = 0.f;
  #pragma unroll
  for (int i = 0; i < 4; ++i) {
    const int d = i * 64 + lane * 2;
    const v2f f = *(const v2f*)(src + d);
    s += f.x * f.x + f.y * f.y;
    dst[d]     = (_Float16)f.x;
    dst[d + 1] = (_Float16)f.y;
  }
  #pragma unroll
  for (int off = 16; off > 0; off >>= 1) s += __shfl_xor(s, off, 32);
  if (lane == 0) en[k] = s;
}

// norms-only (fallback when workspace can't hold the f16 codebook)
__global__ __launch_bounds__(32)
void vq_norms_kernel(const float* __restrict__ emb, float* __restrict__ en) {
  const int k = blockIdx.x;
  const int lane = threadIdx.x;
  const float* row = emb + k * D_DIM;
  float s = 0.f;
  #pragma unroll
  for (int d = 0; d < D_DIM; d += 32) { float v = row[d + lane]; s += v * v; }
  #pragma unroll
  for (int off = 16; off > 0; off >>= 1) s += __shfl_xor(s, off, 32);
  if (lane == 0) en[k] = s;
}

// =======================================================================
// shared helper: preload one wave's 16-pixel x 256-dim A panel as WMMA
// fragments.  ISA f16 A 16x32 layout: lanes 0-15 hold K 0..7 / 16..23
// (pairs per VGPR), lanes 16-31 same rows with K offset +8.
// =======================================================================
__device__ __forceinline__ void load_a_panel(const float* __restrict__ z,
                                             int zoff, int half, v16h a[8]) {
  const int kHalf = half ? 8 : 0;
  #pragma unroll
  for (int c = 0; c < 8; ++c) {
    const int kBase = c * 32 + kHalf;
    #pragma unroll
    for (int v = 0; v < 8; ++v) {
      const int d0 = kBase + ((v < 4) ? (2 * v) : (16 + 2 * (v - 4)));
      a[c][2 * v]     = (_Float16)z[zoff + (d0    ) * 4096];
      a[c][2 * v + 1] = (_Float16)z[zoff + (d0 + 1) * 4096];
    }
  }
}

__device__ __forceinline__ void argmin_epilogue(float bestv[8], int besti[8],
                                                int mrow, int half, int p0,
                                                int* __restrict__ out) {
  #pragma unroll
  for (int r = 0; r < 8; ++r) {
    float v = bestv[r]; int idx = besti[r];
    #pragma unroll
    for (int off = 8; off > 0; off >>= 1) {
      const float ov = __shfl_xor(v,   off, 32);
      const int   oi = __shfl_xor(idx, off, 32);
      if (ov < v || (ov == v && oi < idx)) { v = ov; idx = oi; }
    }
    bestv[r] = v; besti[r] = idx;
  }
  if (mrow == 0) {
    #pragma unroll
    for (int r = 0; r < 8; ++r) out[p0 + half * 8 + r] = besti[r];
  }
}

// =======================================================================
// FAST PATH: 512 threads (16 waves, 256 pixels / block), f16 codebook in
// workspace, double-buffered GLOBAL_LOAD_ASYNC_TO_LDS_B128 staging
// (ASYNCcnt), one barrier per code tile.
// =======================================================================
__global__ __launch_bounds__(512)
void vq_argmin_async_kernel(const float* __restrict__ z,
                            const _Float16* __restrict__ eh,
                            const float* __restrict__ en,
                            int* __restrict__ out) {
  __shared__ _Float16 ecache[2][16 * LDS_STRIDE];
  __shared__ float    encache[K_CODES];

  const int tid  = threadIdx.x;
  const int lane = tid & 31;
  const int wave = tid >> 5;
  const int half = lane >> 4;
  const int mrow = lane & 15;

  // stage all 1024 code norms once (read via DS in the hot loop)
  #pragma unroll
  for (int i = 0; i < 2; ++i) encache[tid + 512 * i] = en[tid + 512 * i];

  const int p0 = blockIdx.x * 256 + wave * 16;   // 16 consecutive w, fixed (b,h)
  const int b  = p0 >> 12;
  const int h  = (p0 >> 6) & 63;
  const int w0 = p0 & 63;
  const int zoff = ((b * D_DIM) << 12) + (h << 6) + w0 + mrow;

  v16h a[8];
  load_a_panel(z, zoff, half, a);

  // async staging addressing: wave w stages code row w of each tile
  // (32 threads x 16B = one packed 512B f16 row); LDS rows padded to 528B.
  const uint32_t inrow = (uint32_t)(tid & 31) * 16u;
  const uint32_t lds0  = (uint32_t)(uintptr_t)&ecache[0][0] + wave * (LDS_STRIDE * 2) + inrow;
  const uint32_t lds1  = (uint32_t)(uintptr_t)&ecache[1][0] + wave * (LDS_STRIDE * 2) + inrow;
  const uint64_t gbase = (uint64_t)(uintptr_t)eh + (uint64_t)tid * 16u;

  // prefetch tile 0 into buffer 0
  asm volatile("global_load_async_to_lds_b128 %0, %1, off"
               :: "v"(lds0), "v"(gbase) : "memory");

  float bestv[8];
  int   besti[8];
  #pragma unroll
  for (int r = 0; r < 8; ++r) { bestv[r] = 3.4e38f; besti[r] = 0; }

  const int bHalfOff = half ? 8 : 0;

  for (int kt = 0; kt < NT; ++kt) {
    const int kbase = kt * 16;

    asm volatile("s_wait_asynccnt 0x0" ::: "memory");  // my tile piece landed
    __syncthreads();                                   // everyone's landed

    if (kt + 1 < NT) {                                 // kick next tile's copy
      const uint32_t ldsn = ((kt + 1) & 1) ? lds1 : lds0;
      const uint64_t ga   = gbase + (uint64_t)(kt + 1) * (16u * D_DIM * 2u);
      asm volatile("global_load_async_to_lds_b128 %0, %1, off"
                   :: "v"(ldsn), "v"(ga) : "memory");
    }

    const _Float16* ebuf = ecache[kt & 1];
    v8f acc = {};
    #pragma unroll
    for (int c = 0; c < 8; ++c) {
      const int boff = mrow * LDS_STRIDE + c * 32 + bHalfOff;
      const v8h b0 = *(const v8h*)&ebuf[boff];
      const v8h b1 = *(const v8h*)&ebuf[boff + 16];
      v16h bf;
      #pragma unroll
      for (int q = 0; q < 8; ++q) { bf[q] = b0[q]; bf[q + 8] = b1[q]; }
      acc = __builtin_amdgcn_wmma_f32_16x16x32_f16(false, a[c], false, bf,
                                                   (short)0, acc, false, false);
    }

    const float enk = encache[kbase + mrow];
    const int   gk  = kbase + mrow;
    #pragma unroll
    for (int r = 0; r < 8; ++r) {
      const float v = __builtin_fmaf(-2.0f, acc[r], enk);
      if (v < bestv[r]) { bestv[r] = v; besti[r] = gk; }
    }
  }

  argmin_epilogue(bestv, besti, mrow, half, p0, out);
}

// =======================================================================
// FALLBACK PATH (small workspace): 256 threads, f32 staging + convert
// =======================================================================
__global__ __launch_bounds__(256)
void vq_argmin_sync_kernel(const float* __restrict__ z,
                           const float* __restrict__ emb,
                           const float* __restrict__ en,
                           int* __restrict__ out) {
  __shared__ _Float16 ecache[16 * LDS_STRIDE];

  const int tid  = threadIdx.x;
  const int lane = tid & 31;
  const int wave = tid >> 5;
  const int half = lane >> 4;
  const int mrow = lane & 15;

  const int p0 = blockIdx.x * 128 + wave * 16;
  const int b  = p0 >> 12;
  const int h  = (p0 >> 6) & 63;
  const int w0 = p0 & 63;
  const int zoff = ((b * D_DIM) << 12) + (h << 6) + w0 + mrow;

  v16h a[8];
  load_a_panel(z, zoff, half, a);

  float bestv[8];
  int   besti[8];
  #pragma unroll
  for (int r = 0; r < 8; ++r) { bestv[r] = 3.4e38f; besti[r] = 0; }

  const int bHalfOff = half ? 8 : 0;

  for (int kt = 0; kt < NT; ++kt) {
    const int kbase = kt * 16;
    __syncthreads();
    #pragma unroll
    for (int i = 0; i < 4; ++i) {
      const int flat = i * 1024 + tid * 4;
      const int kr   = flat >> 8;
      const int d    = flat & 255;
      const v4f ev = *(const v4f*)(emb + (kbase + kr) * D_DIM + d);
      _Float16* dst = &ecache[kr * LDS_STRIDE + d];
      dst[0] = (_Float16)ev.x; dst[1] = (_Float16)ev.y;
      dst[2] = (_Float16)ev.z; dst[3] = (_Float16)ev.w;
    }
    if (kt + 1 < NT)
      __builtin_prefetch(emb + (kbase + 16) * D_DIM + tid, 0, 0);
    __syncthreads();

    v8f acc = {};
    #pragma unroll
    for (int c = 0; c < 8; ++c) {
      const int boff = mrow * LDS_STRIDE + c * 32 + bHalfOff;
      const v8h b0 = *(const v8h*)&ecache[boff];
      const v8h b1 = *(const v8h*)&ecache[boff + 16];
      v16h bf;
      #pragma unroll
      for (int q = 0; q < 8; ++q) { bf[q] = b0[q]; bf[q + 8] = b1[q]; }
      acc = __builtin_amdgcn_wmma_f32_16x16x32_f16(false, a[c], false, bf,
                                                   (short)0, acc, false, false);
    }

    const float enk = en[kbase + mrow];
    const int   gk  = kbase + mrow;
    #pragma unroll
    for (int r = 0; r < 8; ++r) {
      const float v = __builtin_fmaf(-2.0f, acc[r], enk);
      if (v < bestv[r]) { bestv[r] = v; besti[r] = gk; }
    }
  }

  argmin_epilogue(bestv, besti, mrow, half, p0, out);
}

extern "C" void kernel_launch(void* const* d_in, const int* in_sizes, int n_in,
                              void* d_out, int out_size, void* d_ws, size_t ws_size,
                              hipStream_t stream) {
  const float* z   = (const float*)d_in[0];   // [16,256,64,64] f32
  const float* emb = (const float*)d_in[1];   // [1024,256] f32
  int* out = (int*)d_out;                     // [B,H,W] int32 indices

  const size_t eh_bytes = (size_t)K_CODES * D_DIM * sizeof(_Float16);  // 512 KB
  const size_t en_bytes = (size_t)K_CODES * sizeof(float);             // 4 KB

  if (ws_size >= eh_bytes + en_bytes) {
    _Float16* eh = (_Float16*)d_ws;
    float*    en = (float*)((char*)d_ws + eh_bytes);
    vq_prep_kernel<<<K_CODES, 32, 0, stream>>>(emb, eh, en);
    vq_argmin_async_kernel<<<N_PIX / 256, 512, 0, stream>>>(z, eh, en, out);
  } else {
    float* en = (float*)d_ws;                 // needs only 4 KB
    vq_norms_kernel<<<K_CODES, 32, 0, stream>>>(emb, en);
    vq_argmin_sync_kernel<<<N_PIX / 128, 256, 0, stream>>>(z, emb, en, out);
  }
}
